// Capsule_23175643529745
// MI455X (gfx1250) — matrix operand order repaired and checked by reading, not compiled, not run on hardware
//
#include <hip/hip_runtime.h>
#include <cstdint>
#include <cstddef>

// ---------------------------------------------------------------------------
// Capsule routing for MI455X (gfx1250, wave32, WMMA).
// W pre-packed to bf16 (67MB, L2-resident).  Every routing pass recomputes
// u_hat tiles with v_wmma_f32_16x16x32_bf16; two input capsules are packed per
// WMMA K=32.  S-pass: BT=32 (two M-tiles share every B fragment -> W L2
// traffic halved), c folded into A, per-chunk partials stored non-atomically
// and summed by a tiny reduce kernel.  Route-pass: transposed GEMM U' = W x^T
// so the agreement d-reduction is in-lane FMAs + one shfl_xor(16).
// ---------------------------------------------------------------------------

#define BATCH    64
#define IN_CAPS  2048
#define IN_DIM   16
#define NUM_CAPS 32
#define DIM_CAPS 32
#define N_OD     (NUM_CAPS * DIM_CAPS)   // 1024
#define TPB      256                     // 8 wave32 per workgroup

// S-pass tiling: 2 batch-tiles of 16 per WG, i-chunks of 32
#define S_BT     32
#define S_NBT    (BATCH / S_BT)          // 2
#define S_CHUNK  32
#define S_NCH    (IN_CAPS / S_CHUNK)     // 64

// Route-pass tiling: 1 batch-tile of 16 per WG (softmax needs all o in-WG)
#define R_BT     16
#define R_NBT    (BATCH / R_BT)          // 4
#define R_CHUNK  64
#define R_NCH    (IN_CAPS / R_CHUNK)     // 32

typedef __attribute__((ext_vector_type(16))) __bf16 v16bf;
typedef __attribute__((ext_vector_type(8)))  float  v8f;

union FragU {
  v16bf  b;
  __bf16 h[16];
  uint4  q[2];
};

// ---------------------------------------------------------------------------
// One-time f32 -> bf16 pack (native casts -> v_cvt_pk_bf16_f32).
// ---------------------------------------------------------------------------
__global__ void pack_bf16_kernel(const float* __restrict__ src,
                                 __bf16* __restrict__ dst, int n8) {
  const int idx = blockIdx.x * blockDim.x + threadIdx.x;
  if (idx >= n8) return;
  const float4 a = *(const float4*)(src + (size_t)idx * 8);
  const float4 b = *(const float4*)(src + (size_t)idx * 8 + 4);
  union { __bf16 h[8]; uint4 q; } o;
  o.h[0] = (__bf16)a.x; o.h[1] = (__bf16)a.y; o.h[2] = (__bf16)a.z; o.h[3] = (__bf16)a.w;
  o.h[4] = (__bf16)b.x; o.h[5] = (__bf16)b.y; o.h[6] = (__bf16)b.z; o.h[7] = (__bf16)b.w;
  *(uint4*)(dst + (size_t)idx * 8) = o.q;
}

// ---------------------------------------------------------------------------
// S pass: s[b, o*32+d] = sum_i c[b,i,o] * sum_p x[b,i,p] * W[i,o,d,p]
// A = [c_i0*x_i0 | c_i1*x_i1] (K=32), B = [W_i0; W_i1]; each B fragment is
// reused by two M-tiles.  Per-chunk partial written with plain stores.
// ---------------------------------------------------------------------------
__global__ __launch_bounds__(TPB)
void caps_s_kernel(const float* __restrict__ x, const __bf16* __restrict__ Wp,
                   const float* __restrict__ cbuf, float* __restrict__ sparts,
                   int uniform_c) {
  const int wid = threadIdx.x >> 5;          // wave 0..7 -> 128-wide N slice
  const int ln  = threadIdx.x & 15;          // lane%16: A row m / C col n
  const int hf  = (threadIdx.x >> 4) & 1;    // lane half
  const int b0  = blockIdx.y * S_BT;
  const int ibase = blockIdx.x * S_CHUNK;
  const int wbase = wid * 128;
  const int kbase = hf * 8;                  // A half0: p 0..7, half1: p 8..15

  v8f acc[2][8];
#pragma unroll
  for (int mt = 0; mt < 2; ++mt)
#pragma unroll
    for (int t = 0; t < 8; ++t)
#pragma unroll
      for (int r = 0; r < 8; ++r) acc[mt][t][r] = 0.f;

  for (int ii = 0; ii < S_CHUNK; ii += 2) {
    const int i0 = ibase + ii;

    // x rows (f32) for both M-tiles x both capsules of the pair
    float xr[2][2][8];
    float cv[2][2][4];
#pragma unroll
    for (int mt = 0; mt < 2; ++mt) {
      const float* xp0 = x + (((size_t)(b0 + mt * 16 + ln) * IN_CAPS + i0) * IN_DIM + kbase);
      const float4 xa0 = *(const float4*)xp0;
      const float4 xb0 = *(const float4*)(xp0 + 4);
      const float4 xa1 = *(const float4*)(xp0 + IN_DIM);
      const float4 xb1 = *(const float4*)(xp0 + IN_DIM + 4);
      xr[mt][0][0] = xa0.x; xr[mt][0][1] = xa0.y; xr[mt][0][2] = xa0.z; xr[mt][0][3] = xa0.w;
      xr[mt][0][4] = xb0.x; xr[mt][0][5] = xb0.y; xr[mt][0][6] = xb0.z; xr[mt][0][7] = xb0.w;
      xr[mt][1][0] = xa1.x; xr[mt][1][1] = xa1.y; xr[mt][1][2] = xa1.z; xr[mt][1][3] = xa1.w;
      xr[mt][1][4] = xb1.x; xr[mt][1][5] = xb1.y; xr[mt][1][6] = xb1.z; xr[mt][1][7] = xb1.w;
      if (uniform_c) {
#pragma unroll
        for (int q = 0; q < 4; ++q) { cv[mt][0][q] = 1.0f / NUM_CAPS; cv[mt][1][q] = 1.0f / NUM_CAPS; }
      } else {
        const float* cp = cbuf + ((size_t)(b0 + mt * 16 + ln) * IN_CAPS + i0) * NUM_CAPS + wid * 4;
        const float4 c0 = *(const float4*)cp;
        const float4 c1 = *(const float4*)(cp + NUM_CAPS);
        cv[mt][0][0] = c0.x; cv[mt][0][1] = c0.y; cv[mt][0][2] = c0.z; cv[mt][0][3] = c0.w;
        cv[mt][1][0] = c1.x; cv[mt][1][1] = c1.y; cv[mt][1][2] = c1.z; cv[mt][1][3] = c1.w;
      }
    }

    FragU af[2];
#pragma unroll
    for (int t = 0; t < 8; ++t) {
      FragU bfr;                             // every lane loads 16B of i0 and 16B of i1 (p half hf)
      const __bf16* wp = Wp + ((size_t)i0 * N_OD + wbase + t * 16 + ln) * IN_DIM + kbase;
      bfr.q[0] = *(const uint4*)wp;
      bfr.q[1] = *(const uint4*)(wp + (size_t)N_OD * IN_DIM);

      if ((t & 1) == 0) {                    // new o -> rebuild scaled A pair
#pragma unroll
        for (int mt = 0; mt < 2; ++mt) {
          const float s0 = cv[mt][0][t >> 1], s1 = cv[mt][1][t >> 1];
#pragma unroll
          for (int e = 0; e < 8; ++e) af[mt].h[e]     = (__bf16)(xr[mt][0][e] * s0);
#pragma unroll
          for (int e = 0; e < 8; ++e) af[mt].h[8 + e] = (__bf16)(xr[mt][1][e] * s1);
        }
      }
#pragma unroll
      for (int mt = 0; mt < 2; ++mt)
        acc[mt][t] = __builtin_amdgcn_wmma_f32_16x16x32_bf16(
            false, af[mt].b, false, bfr.b, (short)0, acc[mt][t], false, false);
    }
    if (ii + 2 < S_CHUNK)
      __builtin_prefetch(Wp + ((size_t)(i0 + 2 + hf) * N_OD + wbase + ln * 8) * IN_DIM, 0, 1);
  }

  // non-atomic per-chunk partial
  float* outp = sparts + (size_t)blockIdx.x * (BATCH * N_OD);
#pragma unroll
  for (int mt = 0; mt < 2; ++mt)
#pragma unroll
    for (int t = 0; t < 8; ++t) {
      const int od = wbase + t * 16 + ln;
#pragma unroll
      for (int r = 0; r < 8; ++r)
        outp[(size_t)(b0 + mt * 16 + r + 8 * hf) * N_OD + od] = acc[mt][t][r];
    }
}

// ---------------------------------------------------------------------------
// Sum the per-chunk partials: s[idx] = sum_c sparts[c][idx]   (coalesced)
// ---------------------------------------------------------------------------
__global__ void caps_reduce_kernel(const float* __restrict__ sparts,
                                   float* __restrict__ sbuf) {
  const int idx = blockIdx.x * blockDim.x + threadIdx.x;   // < 65536
  float s = 0.f;
  for (int c = 0; c < S_NCH; ++c)
    s += sparts[(size_t)c * (BATCH * N_OD) + idx];
  sbuf[idx] = s;
}

// ---------------------------------------------------------------------------
// Routing pass (transposed): U'[od, b] = W_i x^T.  A0=[W_i0|0], A1=[0|W_i1]
// share one B = x^T (loaded once per i-pair, t-invariant).  C fragment holds
// od in VGPRs and b in lanes, so agreement = sum_d v*U' is in-lane FMAs plus
// a single shfl_xor(16).  Then logits += agreement, c = softmax_o(logits).
// ---------------------------------------------------------------------------
__global__ __launch_bounds__(TPB)
void caps_route_kernel(const __bf16* __restrict__ xp, const __bf16* __restrict__ Wp,
                       const float* __restrict__ vbuf, float* __restrict__ logits,
                       float* __restrict__ cbuf, int first) {
  __shared__ float agg[2][R_BT * NUM_CAPS];  // 4 KB
  const int wid = threadIdx.x >> 5;
  const int ln  = threadIdx.x & 15;          // lane%16: A od-row / C batch col
  const int hf  = (threadIdx.x >> 4) & 1;
  const int b0  = blockIdx.y * R_BT;
  const int ibase = blockIdx.x * R_CHUNK;
  const int wbase = wid * 128;
  const int kbase = hf * 8;

  uint4 z4; z4.x = z4.y = z4.z = z4.w = 0u;

  // vreg[t][r] = v[b0+ln][o(t)][d(t,r)]  (matches transposed C layout)
  float vreg[8][8];
#pragma unroll
  for (int t = 0; t < 8; ++t) {
    const int o = wid * 4 + (t >> 1);
#pragma unroll
    for (int r = 0; r < 8; ++r) {
      const int d = (t & 1) * 16 + r + 8 * hf;
      vreg[t][r] = vbuf[(size_t)(b0 + ln) * N_OD + o * DIM_CAPS + d];
    }
  }

  const int m_sm = threadIdx.x >> 4;   // softmax: 16 lanes per batch row
  const int g_sm = threadIdx.x & 15;

  for (int ii = 0; ii < R_CHUNK; ii += 2) {
    const int i0 = ibase + ii;
    // B = x^T: lane ln covers batch row b0+ln; half selects p 0..7/8..15 of i0 and i1
    FragU bx;
    const __bf16* xrow = xp + ((size_t)(b0 + ln) * IN_CAPS + i0) * IN_DIM + kbase;
    bx.q[0] = *(const uint4*)xrow;
    bx.q[1] = *(const uint4*)(xrow + IN_DIM);

    float ag0 = 0.f, ag1 = 0.f;
#pragma unroll
    for (int t = 0; t < 8; ++t) {
      const int row = wbase + t * 16 + ln;   // od row of this tile
      FragU a0, a1;
      a0.q[1] = z4; a1.q[0] = z4;
      a0.q[0] = *(const uint4*)(Wp + ((size_t)i0 * N_OD + row) * IN_DIM + kbase);
      a1.q[1] = *(const uint4*)(Wp + ((size_t)(i0 + 1) * N_OD + row) * IN_DIM + kbase);

      v8f uz;
#pragma unroll
      for (int r = 0; r < 8; ++r) uz[r] = 0.f;
      const v8f u0 = __builtin_amdgcn_wmma_f32_16x16x32_bf16(
          false, a0.b, false, bx.b, (short)0, uz, false, false);
      const v8f u1 = __builtin_amdgcn_wmma_f32_16x16x32_bf16(
          false, a1.b, false, bx.b, (short)0, uz, false, false);

      if ((t & 1) == 0) { ag0 = 0.f; ag1 = 0.f; }
#pragma unroll
      for (int r = 0; r < 8; ++r) {
        ag0 += u0[r] * vreg[t][r];
        ag1 += u1[r] * vreg[t][r];
      }
      if (t & 1) {                           // o complete: combine row halves
        ag0 += __shfl_xor(ag0, 16, 32);
        ag1 += __shfl_xor(ag1, 16, 32);
        if (hf == 0) {
          const int o = wid * 4 + (t >> 1);
          agg[0][ln * NUM_CAPS + o] = ag0;
          agg[1][ln * NUM_CAPS + o] = ag1;
        }
      }
    }
    __syncthreads();
#pragma unroll
    for (int pi = 0; pi < 2; ++pi) {
      const size_t lidx = ((size_t)(b0 + m_sm) * IN_CAPS + (i0 + pi)) * NUM_CAPS + 2 * g_sm;
      const float a0v = agg[pi][m_sm * NUM_CAPS + 2 * g_sm];
      const float a1v = agg[pi][m_sm * NUM_CAPS + 2 * g_sm + 1];
      float b0v, b1v;
      if (first) { b0v = a0v; b1v = a1v; }
      else       { b0v = logits[lidx] + a0v; b1v = logits[lidx + 1] + a1v; }
      logits[lidx] = b0v; logits[lidx + 1] = b1v;
      float mx = fmaxf(b0v, b1v);
#pragma unroll
      for (int mask = 1; mask < 16; mask <<= 1) mx = fmaxf(mx, __shfl_xor(mx, mask, 32));
      const float e0 = __expf(b0v - mx), e1 = __expf(b1v - mx);
      float sm = e0 + e1;
#pragma unroll
      for (int mask = 1; mask < 16; mask <<= 1) sm += __shfl_xor(sm, mask, 32);
      const float inv = 1.0f / sm;
      cbuf[lidx] = e0 * inv; cbuf[lidx + 1] = e1 * inv;
    }
    __syncthreads();
  }
}

// ---------------------------------------------------------------------------
// squash: v = (|s|^2 / (1+|s|^2)) * s / sqrt(|s|^2 + eps), per (b, o) row.
// ---------------------------------------------------------------------------
__global__ void caps_squash_kernel(const float* __restrict__ s, float* __restrict__ v) {
  const int idx = blockIdx.x * blockDim.x + threadIdx.x;   // b*32 + o
  if (idx >= BATCH * NUM_CAPS) return;
  const float* sp = s + (size_t)idx * DIM_CAPS;
  float sq = 0.f;
#pragma unroll
  for (int d = 0; d < DIM_CAPS; ++d) sq += sp[d] * sp[d];
  const float fac = (sq / (1.f + sq)) * rsqrtf(sq + 1e-7f);
  float* vp = v + (size_t)idx * DIM_CAPS;
#pragma unroll
  for (int d = 0; d < DIM_CAPS; ++d) vp[d] = sp[d] * fac;
}

extern "C" void kernel_launch(void* const* d_in, const int* in_sizes, int n_in,
                              void* d_out, int out_size, void* d_ws, size_t ws_size,
                              hipStream_t stream) {
  (void)in_sizes; (void)n_in; (void)out_size; (void)ws_size;
  const float* x = (const float*)d_in[0];   // [64, 2048, 16]
  const float* W = (const float*)d_in[1];   // [2048, 32, 32, 16]
  float* out = (float*)d_out;               // [64, 32, 32]

  char* ws = (char*)d_ws;
  float*  cbuf   = (float*)ws;                                       // 16 MB
  float*  logits = (float*)(ws + (size_t)16 * 1024 * 1024);          // 16 MB
  float*  sbuf   = (float*)(ws + (size_t)32 * 1024 * 1024);          // 256 KB
  float*  vbuf   = sbuf + (size_t)BATCH * N_OD;                      // 256 KB
  __bf16* xpack  = (__bf16*)(ws + (size_t)33 * 1024 * 1024);         // 4 MB
  __bf16* wpack  = (__bf16*)(ws + (size_t)37 * 1024 * 1024);         // 64 MB
  float*  sparts = (float*)(ws + (size_t)101 * 1024 * 1024);         // 16 MB

  const int nx8 = BATCH * IN_CAPS * IN_DIM / 8;                      // 262144
  const int nw8 = IN_CAPS * N_OD * IN_DIM / 8;                       // 4194304
  pack_bf16_kernel<<<(nx8 + 255) / 256, 256, 0, stream>>>(x, xpack, nx8);
  pack_bf16_kernel<<<(nw8 + 255) / 256, 256, 0, stream>>>(W, wpack, nw8);

  const dim3 gS(S_NCH, S_NBT), gR(R_NCH, R_NBT), blk(TPB);
  const dim3 gRed((BATCH * N_OD + 255) / 256), gSq((BATCH * NUM_CAPS + 255) / 256);

  // iter 1: c uniform (softmax of zeros)
  caps_s_kernel<<<gS, blk, 0, stream>>>(x, wpack, cbuf, sparts, 1);
  caps_reduce_kernel<<<gRed, blk, 0, stream>>>(sparts, sbuf);
  caps_squash_kernel<<<gSq, blk, 0, stream>>>(sbuf, vbuf);
  caps_route_kernel<<<gR, blk, 0, stream>>>(xpack, wpack, vbuf, logits, cbuf, 1);

  // iter 2
  caps_s_kernel<<<gS, blk, 0, stream>>>(x, wpack, cbuf, sparts, 0);
  caps_reduce_kernel<<<gRed, blk, 0, stream>>>(sparts, sbuf);
  caps_squash_kernel<<<gSq, blk, 0, stream>>>(sbuf, vbuf);
  caps_route_kernel<<<gR, blk, 0, stream>>>(xpack, wpack, vbuf, logits, cbuf, 0);

  // iter 3: final s -> squash straight into d_out
  caps_s_kernel<<<gS, blk, 0, stream>>>(x, wpack, cbuf, sparts, 0);
  caps_reduce_kernel<<<gRed, blk, 0, stream>>>(sparts, sbuf);
  caps_squash_kernel<<<gSq, blk, 0, stream>>>(sbuf, out);
}
